// AgentAttention_23356032156343
// MI455X (gfx1250) — matrix operand order repaired and checked by reading, not compiled, not run on hardware
//
#include <hip/hip_runtime.h>
#include <cstdint>
#include <cstddef>

#define DIMC   768
#define HEADS  12
#define HD     64
#define HWD    56
#define NTOK   3136
#define NAG    49
#define NAGP   64
#define NB     16
#define SCALE  0.125f

typedef __attribute__((ext_vector_type(16))) __bf16        v16bf;
typedef __attribute__((ext_vector_type(8)))  float         v8f;
typedef __attribute__((ext_vector_type(8)))  unsigned int  u32x8;
typedef __attribute__((ext_vector_type(4)))  unsigned int  u32x4;

static __device__ __forceinline__ unsigned short f2bf(float f) {
  unsigned u = __float_as_uint(f);
  u += 0x7FFFu + ((u >> 16) & 1u);          // round-to-nearest-even
  return (unsigned short)(u >> 16);
}
static __device__ __forceinline__ float bf2f(unsigned short h) {
  return __uint_as_float(((unsigned)h) << 16);
}

// ---- WMMA fragment loaders (contiguous k = 16*(lane/16) + s in both A and B, wave32) ----------
// A: 16x32 bf16, element (m,k) at p[m*lda + k]; K pairs contiguous -> b32 loads
static __device__ __forceinline__ v16bf load_a_frag(const unsigned short* __restrict__ p,
                                                    int lda, int lane) {
  const int m  = lane & 15;
  const int kb = (lane >> 4) * 16;
  const unsigned short* row = p + (size_t)m * lda;
  u32x8 r;
#pragma unroll
  for (int j = 0; j < 8; ++j) {
    int k = kb + 2 * j;
    r[j] = *(const unsigned int*)(row + k);
  }
  return __builtin_bit_cast(v16bf, r);
}
// B: 32x16 bf16, element (k,n) at p[n*ld + k] (B^T layout, K contiguous)
static __device__ __forceinline__ v16bf load_bt_frag(const unsigned short* __restrict__ p,
                                                     int ld, int lane) {
  const int n  = lane & 15;
  const int kb = (lane >> 4) * 16;
  const unsigned short* col = p + (size_t)n * ld;
  u32x8 r;
#pragma unroll
  for (int j = 0; j < 8; ++j)
    r[j] = *(const unsigned int*)(col + kb + 2 * j);
  return __builtin_bit_cast(v16bf, r);
}

#define WMMA_BF16(Af, Bf, Cf) \
  __builtin_amdgcn_wmma_f32_16x16x32_bf16(false, (Af), false, (Bf), (short)0, (Cf), false, false)

// ---------------------------------------------------------------------------
// 1) split x: cls rows -> d_out (f32 passthrough), body -> bf16 workspace
__global__ __launch_bounds__(256) void k_split_x(const float* __restrict__ x,
                                                 float* __restrict__ out,
                                                 unsigned short* __restrict__ xt_bf) {
  size_t idx = (size_t)blockIdx.x * 256 + threadIdx.x;
  const size_t total = (size_t)NB * 3137 * DIMC;
  if (idx >= total) return;
  int c = (int)(idx % DIMC);
  int t = (int)((idx / DIMC) % 3137);
  int b = (int)(idx / ((size_t)DIMC * 3137));
  float v = x[idx];
  if (t == 0) out[idx] = v;
  else        xt_bf[((size_t)b * NTOK + (t - 1)) * DIMC + c] = f2bf(v);
}

// 2) generic f32 -> bf16
__global__ __launch_bounds__(256) void k_cvt(const float* __restrict__ src,
                                             unsigned short* __restrict__ dst, int n) {
  int i = blockIdx.x * 256 + threadIdx.x;
  if (i < n) dst[i] = f2bf(src[i]);
}

// 3) qkv = xt @ W : M=50176 K=768 N=2304, bf16 out.
//    8 waves / block, block tile 128x64, B staged transposed in LDS.
__global__ __launch_bounds__(256) void k_gemm_qkv(const unsigned short* __restrict__ A,
                                                  const unsigned short* __restrict__ W,
                                                  unsigned short* __restrict__ C) {
  __shared__ __align__(16) unsigned short Bt[64][34];   // [n][k], pad to kill bank conflicts
  const int tid = threadIdx.x;
  const int lane = tid & 31, wv = tid >> 5;
  const int m0 = blockIdx.x * 128 + wv * 16;
  const int n0 = blockIdx.y * 64;
  const int sr = tid >> 3;          // staging k-row 0..31
  const int sc = (tid & 7) * 8;     // staging n-col start
  v8f acc[4] = {};
  for (int k0 = 0; k0 < DIMC; k0 += 32) {
    __syncthreads();
    {
      u32x4 d = *(const u32x4*)(W + (size_t)(k0 + sr) * 2304 + n0 + sc);
#pragma unroll
      for (int j = 0; j < 4; ++j) {
        unsigned v = d[j];
        Bt[sc + 2 * j][sr]     = (unsigned short)v;
        Bt[sc + 2 * j + 1][sr] = (unsigned short)(v >> 16);
      }
    }
    if (k0 + 32 < DIMC)
      __builtin_prefetch(A + (size_t)m0 * DIMC + k0 + 32, 0, 1);
    __syncthreads();
    v16bf af = load_a_frag(A + (size_t)m0 * DIMC + k0, DIMC, lane);
#pragma unroll
    for (int t = 0; t < 4; ++t) {
      v16bf bf = load_bt_frag(&Bt[t * 16][0], 34, lane);
      acc[t] = WMMA_BF16(af, bf, acc[t]);
    }
  }
  const int cn = lane & 15, mb = (lane >> 4) * 8;
#pragma unroll
  for (int t = 0; t < 4; ++t)
#pragma unroll
    for (int r = 0; r < 8; ++r)
      C[(size_t)(m0 + mb + r) * 2304 + n0 + t * 16 + cn] = f2bf(acc[t][r]);
}

// 4) agent pooling: mean over 8x8 q blocks; rows >=49 zeroed. agent (B,64,768) bf16
__global__ __launch_bounds__(256) void k_agent_pool(const unsigned short* __restrict__ qkv,
                                                    unsigned short* __restrict__ agent) {
  int idx = blockIdx.x * 256 + threadIdx.x;
  if (idx >= NB * NAGP * DIMC) return;
  int c = idx % DIMC, a = (idx / DIMC) % NAGP, b = idx / (DIMC * NAGP);
  float s = 0.f;
  if (a < NAG) {
    int i1 = a / 7, j1 = a % 7;
    for (int i2 = 0; i2 < 8; ++i2)
      for (int j2 = 0; j2 < 8; ++j2) {
        int n = (i1 * 8 + i2) * HWD + (j1 * 8 + j2);
        s += bf2f(qkv[((size_t)b * NTOK + n) * 2304 + c]);
      }
    s *= (1.f / 64.f);
  }
  agent[idx] = f2bf(s);
}

// bilinear 7->56, half-pixel (jax.image.resize 'linear' upsample)
static __device__ __forceinline__ float resize7(const float* __restrict__ src, int oy, int ox) {
  float sy = (oy + 0.5f) * 0.125f - 0.5f;
  float sx = (ox + 0.5f) * 0.125f - 0.5f;
  int y0 = (int)floorf(sy); float fy = sy - (float)y0;
  int x0 = (int)floorf(sx); float fx = sx - (float)x0;
  int y0c = min(6, max(0, y0)), y1c = min(6, max(0, y0 + 1));
  int x0c = min(6, max(0, x0)), x1c = min(6, max(0, x0 + 1));
  float v00 = src[y0c * 7 + x0c], v01 = src[y0c * 7 + x1c];
  float v10 = src[y1c * 7 + x0c], v11 = src[y1c * 7 + x1c];
  return v00 * (1.f - fy) * (1.f - fx) + v01 * (1.f - fy) * fx
       + v10 * fy * (1.f - fx)         + v11 * fy * fx;
}

// 5a) bias1[h,a,n] = resize(an_bias)[h,a,n] + ah[h,a,hh] + aw[h,a,ww]
__global__ __launch_bounds__(256) void k_bias1(const float* __restrict__ an,
                                               const float* __restrict__ ah,
                                               const float* __restrict__ aw,
                                               float* __restrict__ bias1) {
  int idx = blockIdx.x * 256 + threadIdx.x;
  if (idx >= HEADS * NAG * NTOK) return;
  int n = idx % NTOK, a = (idx / NTOK) % NAG, h = idx / (NTOK * NAG);
  int hh = n / HWD, ww = n % HWD;
  float v = resize7(an + (size_t)(h * NAG + a) * 49, hh, ww)
          + ah[(size_t)(h * NAG + a) * HWD + hh]
          + aw[(size_t)(h * NAG + a) * HWD + ww];
  bias1[idx] = v;
}
// 5b) bias2[h,n,a] = resize(na_bias)[h,a,n] + ha[h,hh,a] + wa[h,ww,a]
__global__ __launch_bounds__(256) void k_bias2(const float* __restrict__ na,
                                               const float* __restrict__ ha,
                                               const float* __restrict__ wa,
                                               float* __restrict__ bias2) {
  int idx = blockIdx.x * 256 + threadIdx.x;
  if (idx >= HEADS * NTOK * NAG) return;
  int a = idx % NAG, n = (idx / NAG) % NTOK, h = idx / (NAG * NTOK);
  int hh = n / HWD, ww = n % HWD;
  float v = resize7(na + (size_t)(h * NAG + a) * 49, hh, ww)
          + ha[(size_t)(h * HWD + hh) * NAG + a]
          + wa[(size_t)(h * HWD + ww) * NAG + a];
  bias2[idx] = v;
}

// 6) agent scores S[bh,a,n] = scale*(agent @ k^T) + bias1. M=64 K=64 N=3136.
//    k^T is d-contiguous per token -> load_bt straight from global (b128-merged).
__global__ __launch_bounds__(32) void k_agent_scores(const unsigned short* __restrict__ qkv,
                                                     const unsigned short* __restrict__ agent,
                                                     const float* __restrict__ bias1,
                                                     float* __restrict__ S) {
  const int lane = threadIdx.x;
  const int n0 = blockIdx.x * 64, a0 = blockIdx.y * 16, bh = blockIdx.z;
  const int b = bh / HEADS, h = bh % HEADS;
  const unsigned short* Ap = agent + ((size_t)b * NAGP + a0) * DIMC + h * HD;
  const unsigned short* Kp = qkv + (size_t)b * NTOK * 2304 + DIMC + h * HD;  // k part
  v8f acc[4] = {};
#pragma unroll
  for (int k0 = 0; k0 < HD; k0 += 32) {
    v16bf af = load_a_frag(Ap + k0, DIMC, lane);
#pragma unroll
    for (int t = 0; t < 4; ++t) {
      v16bf bf = load_bt_frag(Kp + (size_t)(n0 + t * 16) * 2304 + k0, 2304, lane);
      acc[t] = WMMA_BF16(af, bf, acc[t]);
    }
  }
  const int cn = lane & 15, mb = (lane >> 4) * 8;
  float* Sb = S + (size_t)bh * NAGP * NTOK;
#pragma unroll
  for (int t = 0; t < 4; ++t)
#pragma unroll
    for (int r = 0; r < 8; ++r) {
      int a = a0 + mb + r, n = n0 + t * 16 + cn;
      int ac = a < NAG ? a : NAG - 1;  // padded rows: value irrelevant
      Sb[(size_t)a * NTOK + n] = acc[t][r] * SCALE
                               + bias1[((size_t)h * NAG + ac) * NTOK + n];
    }
}

// 7) row softmax over n=3136 -> P bf16; padded rows -> 0
__global__ __launch_bounds__(256) void k_softmax_agent(const float* __restrict__ S,
                                                       unsigned short* __restrict__ P) {
  const int row = blockIdx.x;         // bh*64 + a
  const int a = row & (NAGP - 1);
  const float* s = S + (size_t)row * NTOK;
  unsigned short* p = P + (size_t)row * NTOK;
  const int t = threadIdx.x;
  if (a >= NAG) { for (int i = t; i < NTOK; i += 256) p[i] = 0; return; }
  __shared__ float red[256];
  float mx = -1e30f;
  for (int i = t; i < NTOK; i += 256) mx = fmaxf(mx, s[i]);
  red[t] = mx; __syncthreads();
  for (int w = 128; w > 0; w >>= 1) { if (t < w) red[t] = fmaxf(red[t], red[t + w]); __syncthreads(); }
  mx = red[0]; __syncthreads();
  float sm = 0.f;
  for (int i = t; i < NTOK; i += 256) sm += __expf(s[i] - mx);
  red[t] = sm; __syncthreads();
  for (int w = 128; w > 0; w >>= 1) { if (t < w) red[t] += red[t + w]; __syncthreads(); }
  float inv = 1.f / red[0];
  for (int i = t; i < NTOK; i += 256) p[i] = f2bf(__expf(s[i] - mx) * inv);
}

// 8) agent_v = P @ v : M=64 K=3136 N=64 per (b,h); 4 waves share staged v chunk.
__global__ __launch_bounds__(128) void k_agent_v(const unsigned short* __restrict__ P,
                                                 const unsigned short* __restrict__ qkv,
                                                 unsigned short* __restrict__ AV) {
  __shared__ __align__(16) unsigned short Vt[64][34];   // [d][k]
  const int tid = threadIdx.x;
  const int lane = tid & 31, wv = tid >> 5;
  const int bh = blockIdx.x, b = bh / HEADS, h = bh % HEADS;
  const int a0 = wv * 16;
  const unsigned short* Ap = P + ((size_t)bh * NAGP + a0) * NTOK;
  const unsigned short* Bp = qkv + (size_t)b * NTOK * 2304 + 2 * DIMC + h * HD;  // v part
  const int sr = tid >> 2;          // token row 0..31
  const int sc = (tid & 3) * 16;    // d col start
  v8f acc[4] = {};
  for (int k0 = 0; k0 < NTOK; k0 += 32) {
    __syncthreads();
#pragma unroll
    for (int half = 0; half < 2; ++half) {
      u32x4 d = *(const u32x4*)(Bp + (size_t)(k0 + sr) * 2304 + sc + half * 8);
#pragma unroll
      for (int j = 0; j < 4; ++j) {
        unsigned v = d[j];
        Vt[sc + half * 8 + 2 * j][sr]     = (unsigned short)v;
        Vt[sc + half * 8 + 2 * j + 1][sr] = (unsigned short)(v >> 16);
      }
    }
    __syncthreads();
    v16bf af = load_a_frag(Ap + k0, NTOK, lane);
#pragma unroll
    for (int t = 0; t < 4; ++t) {
      v16bf bf = load_bt_frag(&Vt[t * 16][0], 34, lane);
      acc[t] = WMMA_BF16(af, bf, acc[t]);
    }
  }
  const int cn = lane & 15, mb = (lane >> 4) * 8;
#pragma unroll
  for (int t = 0; t < 4; ++t)
#pragma unroll
    for (int r = 0; r < 8; ++r)
      AV[((size_t)bh * NAGP + a0 + mb + r) * HD + t * 16 + cn] = f2bf(acc[t][r]);
}

// 9) depthwise 3x3 conv of v image -> f32 buffer (b,n,c)
__global__ __launch_bounds__(256) void k_dwc(const unsigned short* __restrict__ qkv,
                                             const float* __restrict__ w,
                                             const float* __restrict__ bias,
                                             float* __restrict__ out) {
  size_t idx = (size_t)blockIdx.x * 256 + threadIdx.x;
  if (idx >= (size_t)NB * NTOK * DIMC) return;
  int c = (int)(idx % DIMC);
  int n = (int)((idx / DIMC) % NTOK);
  int b = (int)(idx / ((size_t)DIMC * NTOK));
  int hh = n / HWD, ww = n % HWD;
  float s = bias[c];
#pragma unroll
  for (int dy = -1; dy <= 1; ++dy)
#pragma unroll
    for (int dx = -1; dx <= 1; ++dx) {
      int y = hh + dy, x = ww + dx;
      if (y >= 0 && y < HWD && x >= 0 && x < HWD)
        s += bf2f(qkv[((size_t)b * NTOK + y * HWD + x) * 2304 + 2 * DIMC + c])
           * w[c * 9 + (dy + 1) * 3 + (dx + 1)];
    }
  out[idx] = s;
}

// 10) fused q-attention: scores -> softmax(49) -> @agent_v -> +dwc -> bf16
__global__ __launch_bounds__(128) void k_qattn(const unsigned short* __restrict__ qkv,
                                               const unsigned short* __restrict__ agent,
                                               const float* __restrict__ bias2,
                                               const unsigned short* __restrict__ AV,
                                               const float* __restrict__ dwcf,
                                               unsigned short* __restrict__ outbf) {
  __shared__ float sS[4][16][65];
  __shared__ __align__(16) unsigned short sP[4][16][66];
  __shared__ __align__(16) unsigned short AVt[64][66];    // [d][a]
  const int lane = threadIdx.x & 31, wv = threadIdx.x >> 5;
  const int b = blockIdx.z, h = blockIdx.y;
  const int n0 = blockIdx.x * 64 + wv * 16;
  const int cn = lane & 15, mb = (lane >> 4) * 8;

  // stage agent_v (64x64) transposed into LDS (coalesced b128 reads)
  {
    const unsigned short* Bv = AV + (size_t)(b * HEADS + h) * NAGP * HD;
    int a = threadIdx.x >> 1;
    int d0 = (threadIdx.x & 1) * 32;
#pragma unroll
    for (int q8 = 0; q8 < 4; ++q8) {
      u32x4 dd = *(const u32x4*)(Bv + (size_t)a * HD + d0 + q8 * 8);
#pragma unroll
      for (int j = 0; j < 4; ++j) {
        unsigned v = dd[j];
        AVt[d0 + q8 * 8 + 2 * j][a]     = (unsigned short)v;
        AVt[d0 + q8 * 8 + 2 * j + 1][a] = (unsigned short)(v >> 16);
      }
    }
  }

  // scores: q(16x64) @ agent^T(64x64)
  const unsigned short* Ap = qkv + ((size_t)b * NTOK + n0) * 2304 + h * HD;  // q part
  const unsigned short* Bp = agent + (size_t)b * NAGP * DIMC + h * HD;
  v8f acc[4] = {};
#pragma unroll
  for (int k0 = 0; k0 < HD; k0 += 32) {
    v16bf af = load_a_frag(Ap + k0, 2304, lane);
#pragma unroll
    for (int t = 0; t < 4; ++t) {
      v16bf bf = load_bt_frag(Bp + (size_t)(t * 16) * DIMC + k0, DIMC, lane);
      acc[t] = WMMA_BF16(af, bf, acc[t]);
    }
  }
#pragma unroll
  for (int t = 0; t < 4; ++t)
#pragma unroll
    for (int r = 0; r < 8; ++r) {
      int m = mb + r, a = t * 16 + cn;
      float v = acc[t][r] * SCALE;
      if (a < NAG) v += bias2[((size_t)h * NTOK + n0 + m) * NAG + a];
      sS[wv][m][a] = v;
    }
  __syncthreads();
  if (lane < 16) {
    int m = lane;
    float mx = -1e30f;
    for (int a = 0; a < NAG; ++a) mx = fmaxf(mx, sS[wv][m][a]);
    float sm = 0.f;
    for (int a = 0; a < NAG; ++a) sm += __expf(sS[wv][m][a] - mx);
    float inv = 1.f / sm;
    for (int a = 0; a < NAGP; ++a)
      sP[wv][m][a] = (a < NAG) ? f2bf(__expf(sS[wv][m][a] - mx) * inv) : (unsigned short)0;
  }
  __syncthreads();
  // out = P(16x64) @ agent_v(64x64), B from LDS (transposed)
  v8f o[4] = {};
#pragma unroll
  for (int k0 = 0; k0 < NAGP; k0 += 32) {
    v16bf af = load_a_frag(&sP[wv][0][0] + k0, 66, lane);
#pragma unroll
    for (int t = 0; t < 4; ++t) {
      v16bf bf = load_bt_frag(&AVt[t * 16][0] + k0, 66, lane);
      o[t] = WMMA_BF16(af, bf, o[t]);
    }
  }
#pragma unroll
  for (int t = 0; t < 4; ++t)
#pragma unroll
    for (int r = 0; r < 8; ++r) {
      int m = mb + r, d = t * 16 + cn;
      size_t gi = ((size_t)b * NTOK + n0 + m) * DIMC + h * HD + d;
      outbf[gi] = f2bf(o[t][r] + dwcf[gi]);
    }
}

// 11) final = sum @ proj_w + proj_b -> d_out rows 1..3136 (f32)
__global__ __launch_bounds__(256) void k_gemm_proj(const unsigned short* __restrict__ A,
                                                   const unsigned short* __restrict__ W,
                                                   const float* __restrict__ bias,
                                                   float* __restrict__ out) {
  __shared__ __align__(16) unsigned short Bt[64][34];
  const int tid = threadIdx.x;
  const int lane = tid & 31, wv = tid >> 5;
  const int m0 = blockIdx.x * 128 + wv * 16;
  const int n0 = blockIdx.y * 64;
  const int sr = tid >> 3;
  const int sc = (tid & 7) * 8;
  v8f acc[4] = {};
  for (int k0 = 0; k0 < DIMC; k0 += 32) {
    __syncthreads();
    {
      u32x4 d = *(const u32x4*)(W + (size_t)(k0 + sr) * DIMC + n0 + sc);
#pragma unroll
      for (int j = 0; j < 4; ++j) {
        unsigned v = d[j];
        Bt[sc + 2 * j][sr]     = (unsigned short)v;
        Bt[sc + 2 * j + 1][sr] = (unsigned short)(v >> 16);
      }
    }
    if (k0 + 32 < DIMC)
      __builtin_prefetch(A + (size_t)m0 * DIMC + k0 + 32, 0, 1);
    __syncthreads();
    v16bf af = load_a_frag(A + (size_t)m0 * DIMC + k0, DIMC, lane);
#pragma unroll
    for (int t = 0; t < 4; ++t) {
      v16bf bf = load_bt_frag(&Bt[t * 16][0], 34, lane);
      acc[t] = WMMA_BF16(af, bf, acc[t]);
    }
  }
  const int cn = lane & 15, mb = (lane >> 4) * 8;
#pragma unroll
  for (int t = 0; t < 4; ++t)
#pragma unroll
    for (int r = 0; r < 8; ++r) {
      int m = m0 + mb + r;
      int b = m / NTOK, n = m % NTOK, col = n0 + t * 16 + cn;
      out[((size_t)b * 3137 + 1 + n) * DIMC + col] = acc[t][r] + bias[col];
    }
}

// ---------------------------------------------------------------------------
extern "C" void kernel_launch(void* const* d_in, const int* in_sizes, int n_in,
                              void* d_out, int out_size, void* d_ws, size_t ws_size,
                              hipStream_t stream) {
  (void)in_sizes; (void)n_in; (void)out_size; (void)ws_size;
  const float* x      = (const float*)d_in[0];
  const float* qkv_w  = (const float*)d_in[1];
  const float* proj_w = (const float*)d_in[2];
  const float* proj_b = (const float*)d_in[3];
  const float* dwc_w  = (const float*)d_in[4];
  const float* dwc_b  = (const float*)d_in[5];
  const float* an_b   = (const float*)d_in[6];
  const float* ah_b   = (const float*)d_in[7];
  const float* aw_b   = (const float*)d_in[8];
  const float* na_b   = (const float*)d_in[9];
  const float* ha_b   = (const float*)d_in[10];
  const float* wa_b   = (const float*)d_in[11];
  float* out = (float*)d_out;

  char* ws = (char*)d_ws;
  size_t off = 0;
  auto take = [&](size_t bytes) -> char* {
    char* p = ws + off; off += (bytes + 255) & ~(size_t)255; return p;
  };
  unsigned short* qkv_bf  = (unsigned short*)take((size_t)NB * NTOK * 2304 * 2);
  unsigned short* xt_bf   = (unsigned short*)take((size_t)NB * NTOK * DIMC * 2);
  unsigned short* wq_bf   = (unsigned short*)take((size_t)DIMC * 2304 * 2);
  unsigned short* wp_bf   = (unsigned short*)take((size_t)DIMC * DIMC * 2);
  unsigned short* agent_bf= (unsigned short*)take((size_t)NB * NAGP * DIMC * 2);
  float*          bias1   = (float*)take((size_t)HEADS * NAG * NTOK * 4);
  float*          bias2   = (float*)take((size_t)HEADS * NTOK * NAG * 4);
  float*          Sbuf    = (float*)take((size_t)NB * HEADS * NAGP * NTOK * 4);          // reused: dwc f32
  unsigned short* Pbuf    = (unsigned short*)take((size_t)NB * HEADS * NAGP * NTOK * 2); // reused: sum bf16
  unsigned short* AV      = (unsigned short*)take((size_t)NB * HEADS * NAGP * HD * 2);
  float*          dwcf    = Sbuf;    // same size: 16*3136*768*4 == 192*64*3136*4
  unsigned short* sum_bf  = Pbuf;    // same size: 16*3136*768*2 == 192*64*3136*2

  const size_t tot_x = (size_t)NB * 3137 * DIMC;
  k_split_x<<<dim3((unsigned)((tot_x + 255) / 256)), 256, 0, stream>>>(x, out, xt_bf);
  k_cvt<<<dim3((DIMC * 2304 + 255) / 256), 256, 0, stream>>>(qkv_w, wq_bf, DIMC * 2304);
  k_cvt<<<dim3((DIMC * DIMC + 255) / 256), 256, 0, stream>>>(proj_w, wp_bf, DIMC * DIMC);

  k_gemm_qkv<<<dim3(NB * NTOK / 128, 2304 / 64), 256, 0, stream>>>(xt_bf, wq_bf, qkv_bf);

  k_agent_pool<<<dim3((NB * NAGP * DIMC + 255) / 256), 256, 0, stream>>>(qkv_bf, agent_bf);
  k_bias1<<<dim3((HEADS * NAG * NTOK + 255) / 256), 256, 0, stream>>>(an_b, ah_b, aw_b, bias1);
  k_bias2<<<dim3((HEADS * NTOK * NAG + 255) / 256), 256, 0, stream>>>(na_b, ha_b, wa_b, bias2);

  k_agent_scores<<<dim3(NTOK / 64, NAGP / 16, NB * HEADS), 32, 0, stream>>>(qkv_bf, agent_bf, bias1, Sbuf);
  k_softmax_agent<<<dim3(NB * HEADS * NAGP), 256, 0, stream>>>(Sbuf, Pbuf);
  k_agent_v<<<dim3(NB * HEADS), 128, 0, stream>>>(Pbuf, qkv_bf, AV);

  // S no longer needed -> reuse as dwc f32 buffer
  k_dwc<<<dim3((unsigned)(((size_t)NB * NTOK * DIMC + 255) / 256)), 256, 0, stream>>>(qkv_bf, dwc_w, dwc_b, dwcf);

  // P no longer needed -> reuse as bf16 sum buffer
  k_qattn<<<dim3(NTOK / 64, HEADS, NB), 128, 0, stream>>>(qkv_bf, agent_bf, bias2, AV, dwcf, sum_bf);

  k_gemm_proj<<<dim3(NB * NTOK / 128, DIMC / 64), 256, 0, stream>>>(sum_bf, wp_bf, proj_b, out);
}
